// Binarization_17600775979088
// MI455X (gfx1250) — compile-verified, matrix operand order checked by reading, not run on hardware
//
#include <hip/hip_runtime.h>

typedef __attribute__((ext_vector_type(8))) int v8i;

#define BATCHES 16
#define HDIM 1024
#define WDIM 1024
#define HW (HDIM * WDIM)                  // 1<<20
#define CHUNK 8192
#define BLOCKS_PER_BATCH (HW / CHUNK)     // 128
#define NBLOCKS (BATCHES * BLOCKS_PER_BATCH) // 2048
#define THREADS 256
#define PIX_PER_THREAD (CHUNK / THREADS)  // 32

// ---------------------------------------------------------------------------
// Kernel 1: indicator write + per-block foreground count.
// Count reduction uses V_WMMA_I32_16X16X64_IU8: A = 16x64 mask bytes striped
// across the wave (32 bytes/lane), B = all-ones bytes, C = 0. Each D element
// is a row-sum of A; summing the 8 accumulator VGPRs per lane and combining
// lane 0 / lane 16 of each wave yields the full 1024-byte sum per wave.
// ---------------------------------------------------------------------------
__global__ void __launch_bounds__(THREADS)
indicator_count_kernel(const float* __restrict__ maps,
                       const float* __restrict__ alpha,
                       float* __restrict__ indicator,
                       int* __restrict__ blockCounts) {
    const int blk = blockIdx.x;
    const int tid = threadIdx.x;
    const long base = (long)blk * CHUNK + (long)tid * PIX_PER_THREAD;
    const float thr = 255.0f * alpha[0];

    const float4* __restrict__ src = (const float4*)(maps + base);
    float4* __restrict__ dst = (float4*)(indicator + base);

    __builtin_prefetch((const void*)src, 0, 1);  // global_prefetch_b8

    v8i a;
#pragma unroll
    for (int q = 0; q < 8; ++q) {
        float4 p = src[q];
        unsigned m0 = (p.x < thr) ? 1u : 0u;
        unsigned m1 = (p.y < thr) ? 1u : 0u;
        unsigned m2 = (p.z < thr) ? 1u : 0u;
        unsigned m3 = (p.w < thr) ? 1u : 0u;
        a[q] = (int)(m0 | (m1 << 8) | (m2 << 16) | (m3 << 24));
        float4 o;
        o.x = (float)m0; o.y = (float)m1; o.z = (float)m2; o.w = (float)m3;
        dst[q] = o;
    }

    v8i b;
#pragma unroll
    for (int q = 0; q < 8; ++q) b[q] = 0x01010101;  // all-ones bytes
    v8i c = {};

    // D = A(16x64 u8) x B(64x16 u8) + 0  -> every D element is a row-sum of A
    v8i d = __builtin_amdgcn_wmma_i32_16x16x64_iu8(
        /*sgn_a=*/false, a, /*sgn_b=*/false, b, c,
        /*reuse_a=*/false, /*reuse_b=*/false);

    int s = d[0] + d[1] + d[2] + d[3] + d[4] + d[5] + d[6] + d[7];

    __shared__ int red[THREADS];
    red[tid] = s;
    __syncthreads();
    if (tid == 0) {
        int tot = 0;
#pragma unroll
        for (int w = 0; w < THREADS / 32; ++w)
            tot += red[w * 32] + red[w * 32 + 16];  // rows 0-7 + rows 8-15 per wave
        blockCounts[blk] = tot;
    }
}

// ---------------------------------------------------------------------------
// Kernel 2: per-batch exclusive scan of 128 block counts (one block per batch)
// ---------------------------------------------------------------------------
__global__ void __launch_bounds__(BLOCKS_PER_BATCH)
scan_kernel(const int* __restrict__ blockCounts,
            int* __restrict__ blockOffsets,
            int* __restrict__ countsInt,
            float* __restrict__ countsOut) {
    __shared__ int s[BLOCKS_PER_BATCH];
    const int b = blockIdx.x;
    const int t = threadIdx.x;
    const int v = blockCounts[b * BLOCKS_PER_BATCH + t];
    s[t] = v;
    __syncthreads();
#pragma unroll
    for (int off = 1; off < BLOCKS_PER_BATCH; off <<= 1) {
        int add = (t >= off) ? s[t - off] : 0;
        __syncthreads();
        s[t] += add;
        __syncthreads();
    }
    blockOffsets[b * BLOCKS_PER_BATCH + t] = s[t] - v;  // exclusive
    if (t == BLOCKS_PER_BATCH - 1) {
        countsInt[b] = s[t];
        countsOut[b] = (float)s[t];
    }
}

// ---------------------------------------------------------------------------
// Kernel 3: rank-and-scatter coords. Thread-contiguous pixel mapping + LDS
// scan over tid preserves row-major stable ordering exactly.
// ---------------------------------------------------------------------------
__global__ void __launch_bounds__(THREADS)
scatter_kernel(const float* __restrict__ indicator,
               const int* __restrict__ blockOffsets,
               float* __restrict__ coords) {
    const int blk = blockIdx.x;
    const int tid = threadIdx.x;
    const int batch = blk / BLOCKS_PER_BATCH;
    const long base = (long)blk * CHUNK + (long)tid * PIX_PER_THREAD;

    const float4* __restrict__ src = (const float4*)(indicator + base);
    __builtin_prefetch((const void*)src, 0, 1);

    float m[PIX_PER_THREAD];
    int c = 0;
#pragma unroll
    for (int q = 0; q < 8; ++q) {
        float4 p = src[q];
        m[4 * q + 0] = p.x; m[4 * q + 1] = p.y;
        m[4 * q + 2] = p.z; m[4 * q + 3] = p.w;
        c += (p.x > 0.5f) + (p.y > 0.5f) + (p.z > 0.5f) + (p.w > 0.5f);
    }

    __shared__ int s[THREADS];
    s[tid] = c;
    __syncthreads();
#pragma unroll
    for (int off = 1; off < THREADS; off <<= 1) {
        int add = (tid >= off) ? s[tid - off] : 0;
        __syncthreads();
        s[tid] += add;
        __syncthreads();
    }
    int pos = blockOffsets[blk] + (s[tid] - c);

    const long flatBase = base - (long)batch * HW;  // flat index within batch
    float* __restrict__ cbase = coords + (long)batch * HW * 2;
#pragma unroll
    for (int k = 0; k < PIX_PER_THREAD; ++k) {
        if (m[k] > 0.5f) {
            long flat = flatBase + k;
            int i = (int)(flat >> 10);        // W == 1024
            int j = (int)(flat & (WDIM - 1));
            cbase[(long)pos * 2 + 0] = (float)i;
            cbase[(long)pos * 2 + 1] = (float)j;
            ++pos;
        }
    }
}

// ---------------------------------------------------------------------------
// Kernel 4: pad coords tail [count, HW) with -1
// ---------------------------------------------------------------------------
__global__ void __launch_bounds__(THREADS)
pad_kernel(const int* __restrict__ countsInt, float* __restrict__ coords) {
    const unsigned t = blockIdx.x * THREADS + threadIdx.x;  // < 16M
    const int b = t >> 20;
    const int p = t & (HW - 1);
    if (p >= countsInt[b]) {
        float* c = coords + ((long)b * HW + p) * 2;
        c[0] = -1.0f;
        c[1] = -1.0f;
    }
}

extern "C" void kernel_launch(void* const* d_in, const int* in_sizes, int n_in,
                              void* d_out, int out_size, void* d_ws, size_t ws_size,
                              hipStream_t stream) {
    const float* maps  = (const float*)d_in[0];   // [16,1024,1024] f32
    const float* alpha = (const float*)d_in[1];   // [1] f32

    float* out       = (float*)d_out;
    float* indicator = out;                                   // 16M floats
    float* coords    = out + (size_t)BATCHES * HW;            // 32M floats
    float* countsOut = out + (size_t)BATCHES * HW * 3;        // 16 floats

    int* blockCounts  = (int*)d_ws;                // [2048]
    int* blockOffsets = blockCounts + NBLOCKS;     // [2048]
    int* countsInt    = blockOffsets + NBLOCKS;    // [16]

    indicator_count_kernel<<<NBLOCKS, THREADS, 0, stream>>>(
        maps, alpha, indicator, blockCounts);
    scan_kernel<<<BATCHES, BLOCKS_PER_BATCH, 0, stream>>>(
        blockCounts, blockOffsets, countsInt, countsOut);
    scatter_kernel<<<NBLOCKS, THREADS, 0, stream>>>(
        indicator, blockOffsets, coords);
    pad_kernel<<<(BATCHES * HW) / THREADS, THREADS, 0, stream>>>(
        countsInt, coords);
}
